// PT_GRUMB_52355651338918
// MI455X (gfx1250) — matrix operand order repaired and checked by reading, not compiled, not run on hardware
//
#include <hip/hip_runtime.h>
#include <hip/hip_bf16.h>

// ---------------------------------------------------------------------------
// GRU-memory-block forward, MI455X (gfx1250), wave32 + v_wmma_f32_16x16x32_bf16
// + Tensor Data Mover (tensor_load_to_lds) double-buffered staging.
//
// Live computation (write-gate path is dead code in the reference):
//   G1 = sigmoid(w_inp@x + w_rec_inp@out0)
//   G2 = sigmoid(w_inpgate@x + w_mem_inpgate@mem0 + w_rec_inpgate@out0)
//   G3 = sigmoid(w_readgate@x + w_rec_readgate@out0 + w_mem_readgate@mem0)
//   hidden = G3 + G1*G2
//   out = tanh(w_hid_out @ hidden)
//
// Pipeline:
//   1) cvt kernels: weights f32->bf16 (row-major); activations f32->bf16
//      *transposed* (Xᵀ [B,K]) so GEMM B-fragments are K-contiguous per lane.
//   2) gates kernel: fused 8-GEMM accumulation over K segments {x,mem0,out0},
//      TDM (or coop) LDS staging, sigmoid epilogue -> bf16 hiddenᵀ (packed).
//   3) out kernel: w_hid_out @ hidden, tanh epilogue, f32 out.
// ---------------------------------------------------------------------------

#define MSZ 1024
#define BSZ 4096
#define LDS_PITCH 40        // 32 K-elems + 8 pad elems (80 B rows, 16B aligned)
#define TILE_ELEMS (64 * LDS_PITCH)

typedef __attribute__((ext_vector_type(16))) __bf16 v16bf;
typedef __attribute__((ext_vector_type(8)))  float  v8f;
typedef unsigned int u32x4 __attribute__((ext_vector_type(4)));
typedef int          i32x4 __attribute__((ext_vector_type(4)));
typedef int          i32x8 __attribute__((ext_vector_type(8)));

union Frag { v16bf v; uint4 q[2]; };

#if defined(__gfx1250__) && __has_builtin(__builtin_amdgcn_tensor_load_to_lds) && \
    __has_builtin(__builtin_amdgcn_s_wait_tensorcnt)
#define USE_TDM 1
#else
#define USE_TDM 0
#endif

__device__ __forceinline__ v8f wmma_bf16(v16bf a, v16bf b, v8f c) {
  return __builtin_amdgcn_wmma_f32_16x16x32_bf16(
      /*neg_a=*/false, a, /*neg_b=*/false, b,
      /*c_mod=*/(short)0, c, /*reuse_a=*/false, /*reuse_b=*/false);
}

__device__ __forceinline__ unsigned short f32_to_bf16(float f) {
  union { float f; unsigned u; } cv; cv.f = f;
  unsigned u = cv.u;
  u += 0x7FFFu + ((u >> 16) & 1u);   // round-to-nearest-even
  return (unsigned short)(u >> 16);
}

__device__ __forceinline__ float sigmoidf_fast(float x) {
  return 1.0f / (1.0f + __expf(-x));
}

__device__ __forceinline__ float tanhf_fast(float x) {
  float e = __expf(2.0f * x);
  return (e - 1.0f) / (e + 1.0f);
}

#if USE_TDM
// Stage one [64 B-rows x 32 K-elems] bf16 tile from XT (row stride MSZ elems)
// into LDS with pad: 4 DWORDs pad per 16 DWORDs -> pitch-40 ushort rows.
__device__ __forceinline__ void tdm_stage(const unsigned short* gsrc,
                                          unsigned short* ldst) {
  unsigned long long ga = (unsigned long long)gsrc;
  u32x4 g0;
  g0.x = 1u;                                                  // count=1 (valid user D#)
  g0.y = (unsigned)(unsigned long long)ldst;                  // lds_addr = addr[31:0]
  g0.z = (unsigned)ga;                                        // global_addr[31:0]
  g0.w = (unsigned)((ga >> 32) & 0x01FFFFFFu) | (2u << 30);   // addr[56:32] | type=2
  i32x8 g1;
  g1[0] = (int)((1u << 16)    // data_size = 2 bytes
              | (1u << 20)    // pad_enable
              | (3u << 22)    // pad_interval: every 16 DWORDs
              | (3u << 25));  // pad_amount: 4 DWORDs
  g1[1] = (int)(((unsigned)MSZ & 0xFFFFu) << 16);             // tensor_dim0 lo16
  g1[2] = (int)(((unsigned)MSZ >> 16) |
                (((unsigned)BSZ & 0xFFFFu) << 16));           // dim0 hi | dim1 lo
  g1[3] = (int)(((unsigned)BSZ >> 16) | (32u << 16));         // dim1 hi | tile_dim0=32
  g1[4] = 64;                                                 // tile_dim1=64 (tile_dim2=0)
  g1[5] = (int)MSZ;                                           // tensor_dim0_stride lo32
  g1[6] = 0;
  g1[7] = 0;
  i32x4 gz = {0, 0, 0, 0};
#if __clang_major__ >= 23
  i32x8 gz8 = {0, 0, 0, 0, 0, 0, 0, 0};
  __builtin_amdgcn_tensor_load_to_lds(g0, g1, gz, gz, gz8, 0);
#else
  __builtin_amdgcn_tensor_load_to_lds(g0, g1, gz, gz, 0);
#endif
}
#endif

// One K-segment (1024) of the fused gate GEMMs. XT is [BSZ, MSZ] bf16
// transposed activations; A* are [MSZ, MSZ] bf16 row-major weights.
template <bool HAS_G1>
__device__ __forceinline__ void seg_gemm(
    const unsigned short* __restrict__ XT,
    const unsigned short* __restrict__ A1,
    const unsigned short* __restrict__ A2,
    const unsigned short* __restrict__ A3,
    unsigned short* lds,                 // [2][TILE_ELEMS]
    v8f acc1[4], v8f acc2[4], v8f acc3[4],
    int n0, int mA, int hiK, int lane, int wave, int tid) {
#if USE_TDM
  if (wave == 0) tdm_stage(XT + (size_t)n0 * MSZ, lds);
#pragma unroll 2
  for (int t = 0; t < 32; ++t) {
    const int k0 = t * 32;
    unsigned short* cur = lds + (t & 1) * TILE_ELEMS;
    if (wave == 0) {
      if (t + 1 < 32) {
        tdm_stage(XT + (size_t)n0 * MSZ + (k0 + 32),
                  lds + ((t + 1) & 1) * TILE_ELEMS);
        __builtin_amdgcn_s_wait_tensorcnt(1);   // current tile landed
      } else {
        __builtin_amdgcn_s_wait_tensorcnt(0);   // drain
      }
    }
    __syncthreads();                            // tile t visible to all waves
#else
  for (int t = 0; t < 32; ++t) {
    const int k0 = t * 32;
    unsigned short* cur = lds;
    __syncthreads();
    {
      // Cooperative copy: contiguous b128 global load -> contiguous b128
      // LDS store (no transpose needed: XT is already K-contiguous).
      const int nrow = tid >> 2;
      const int ch   = (tid & 3) * 8;
      uint4 d = *(const uint4*)(XT + (size_t)(n0 + nrow) * MSZ + k0 + ch);
      *(uint4*)(cur + nrow * LDS_PITCH + ch) = d;
    }
    __syncthreads();
#endif
    // A fragments (16x32 bf16): lane<16 holds K{0..7,16..23}, lane>=16 holds
    // K{8..15,24..31} of weight row mA. Weights are L2-resident (18 MB).
    const size_t arow = (size_t)mA * MSZ + (size_t)k0 + (hiK ? 8 : 0);
    Frag a1 = {}, a2 = {}, a3 = {};
    if (HAS_G1) {
      a1.q[0] = *(const uint4*)(A1 + arow);
      a1.q[1] = *(const uint4*)(A1 + arow + 16);
    }
    a2.q[0] = *(const uint4*)(A2 + arow);
    a2.q[1] = *(const uint4*)(A2 + arow + 16);
    a3.q[0] = *(const uint4*)(A3 + arow);
    a3.q[1] = *(const uint4*)(A3 + arow + 16);

#pragma unroll
    for (int j = 0; j < 4; ++j) {
      const unsigned short* bp =
          cur + (j * 16 + (lane & 15)) * LDS_PITCH + (hiK ? 8 : 0);
      Frag b;
      b.q[0] = *(const uint4*)(bp);
      b.q[1] = *(const uint4*)(bp + 16);
      if (HAS_G1) acc1[j] = wmma_bf16(a1.v, b.v, acc1[j]);
      acc2[j] = wmma_bf16(a2.v, b.v, acc2[j]);
      acc3[j] = wmma_bf16(a3.v, b.v, acc3[j]);
    }
#if USE_TDM
    __syncthreads();                            // all reads done before overwrite
#endif
  }
}

// ---------------------------------------------------------------------------
// f32 -> bf16 conversion (row-major kept), 4 elements / thread — weights.
// ---------------------------------------------------------------------------
__global__ __launch_bounds__(256) void PT_GRUMB_cvt(
    const float* __restrict__ s, unsigned short* __restrict__ d, int n4) {
  int i = blockIdx.x * blockDim.x + threadIdx.x;
  if (i < n4) {
    float4 f = ((const float4*)s)[i];
    uint2 o;
    o.x = (unsigned)f32_to_bf16(f.x) | ((unsigned)f32_to_bf16(f.y) << 16);
    o.y = (unsigned)f32_to_bf16(f.z) | ((unsigned)f32_to_bf16(f.w) << 16);
    ((uint2*)d)[i] = o;
  }
}

// ---------------------------------------------------------------------------
// f32 [K=1024, N=4096] -> bf16 transposed [N=4096, K=1024], 32x32 LDS tiles.
// ---------------------------------------------------------------------------
__global__ __launch_bounds__(256) void PT_GRUMB_cvt_t(
    const float* __restrict__ s, unsigned short* __restrict__ d) {
  __shared__ unsigned short t[32 * 33];
  const int tx = threadIdx.x;        // 0..31
  const int ty = threadIdx.y;        // 0..7
  const int n0 = blockIdx.x * 32;
  const int k0 = blockIdx.y * 32;
#pragma unroll
  for (int i = 0; i < 4; ++i) {
    const int k = ty + i * 8;
    t[tx * 33 + k] = f32_to_bf16(s[(size_t)(k0 + k) * BSZ + n0 + tx]);
  }
  __syncthreads();
#pragma unroll
  for (int i = 0; i < 4; ++i) {
    const int n = ty + i * 8;
    d[(size_t)(n0 + n) * MSZ + k0 + tx] = t[n * 33 + tx];
  }
}

// ---------------------------------------------------------------------------
// Fused gate GEMMs. Block = 256 threads = 8 waves; block tile M128 x B64.
// Each wave: 16-row M strip, 3 gates x 4 N-subtiles of 16x16 f32 accum.
// Epilogue writes bf16 hiddenᵀ [B, M] with packed b128 stores.
// ---------------------------------------------------------------------------
__global__ __launch_bounds__(256, 2) void PT_GRUMB_gates(
    const unsigned short* __restrict__ xT,
    const unsigned short* __restrict__ mem0T,
    const unsigned short* __restrict__ out0T,
    const unsigned short* __restrict__ w_inp_b,
    const unsigned short* __restrict__ w_inpgate_b,
    const unsigned short* __restrict__ w_readgate_b,
    const unsigned short* __restrict__ w_mem_inpgate_b,
    const unsigned short* __restrict__ w_mem_readgate_b,
    const unsigned short* __restrict__ w_rec_inp_b,
    const unsigned short* __restrict__ w_rec_inpgate_b,
    const unsigned short* __restrict__ w_rec_readgate_b,
    unsigned short* __restrict__ hiddenT) {
  __shared__ unsigned short ldsX[2 * TILE_ELEMS];

  const int tid  = threadIdx.x;
  const int lane = tid & 31;
  const int wave = tid >> 5;
  const int n0   = blockIdx.x * 64;
  const int m16  = blockIdx.y * 128 + wave * 16;
  const int mA   = m16 + (lane & 15);
  const int hiK  = lane >> 4;

  v8f zero = {0.f, 0.f, 0.f, 0.f, 0.f, 0.f, 0.f, 0.f};
  v8f acc1[4], acc2[4], acc3[4];
#pragma unroll
  for (int j = 0; j < 4; ++j) { acc1[j] = zero; acc2[j] = zero; acc3[j] = zero; }

  // Segment 0: x     -> G1 += w_inp,      G2 += w_inpgate,     G3 += w_readgate
  seg_gemm<true >(xT,    w_inp_b,     w_inpgate_b,     w_readgate_b,
                  ldsX, acc1, acc2, acc3, n0, mA, hiK, lane, wave, tid);
  // Segment 1: mem0  ->                   G2 += w_mem_inpgate, G3 += w_mem_readgate
  seg_gemm<false>(mem0T, nullptr,     w_mem_inpgate_b, w_mem_readgate_b,
                  ldsX, acc1, acc2, acc3, n0, mA, hiK, lane, wave, tid);
  // Segment 2: out0  -> G1 += w_rec_inp,  G2 += w_rec_inpgate, G3 += w_rec_readgate
  seg_gemm<true >(out0T, w_rec_inp_b, w_rec_inpgate_b, w_rec_readgate_b,
                  ldsX, acc1, acc2, acc3, n0, mA, hiK, lane, wave, tid);

  // Epilogue: hidden = sigmoid(G3) + sigmoid(G1)*sigmoid(G2); packed bf16
  // stores, transposed layout hiddenᵀ[n, m] (8 consecutive m per lane).
#pragma unroll
  for (int j = 0; j < 4; ++j) {
    const int n = n0 + j * 16 + (lane & 15);
    unsigned p[4];
#pragma unroll
    for (int r = 0; r < 4; ++r) {
      float hA = sigmoidf_fast(acc3[j][2 * r]) +
                 sigmoidf_fast(acc1[j][2 * r]) * sigmoidf_fast(acc2[j][2 * r]);
      float hB = sigmoidf_fast(acc3[j][2 * r + 1]) +
                 sigmoidf_fast(acc1[j][2 * r + 1]) * sigmoidf_fast(acc2[j][2 * r + 1]);
      p[r] = (unsigned)f32_to_bf16(hA) | ((unsigned)f32_to_bf16(hB) << 16);
    }
    uint4 o; o.x = p[0]; o.y = p[1]; o.z = p[2]; o.w = p[3];
    *(uint4*)(hiddenT + (size_t)n * MSZ + m16 + hiK * 8) = o;
  }
}

// ---------------------------------------------------------------------------
// out = tanh(w_hid_out @ hidden): same staging, single accumulator set.
// ---------------------------------------------------------------------------
__global__ __launch_bounds__(256, 2) void PT_GRUMB_out(
    const unsigned short* __restrict__ Wb,
    const unsigned short* __restrict__ HT,
    float* __restrict__ out) {
  __shared__ unsigned short ldsX[2 * TILE_ELEMS];

  const int tid  = threadIdx.x;
  const int lane = tid & 31;
  const int wave = tid >> 5;
  const int n0   = blockIdx.x * 64;
  const int m16  = blockIdx.y * 128 + wave * 16;
  const int mA   = m16 + (lane & 15);
  const int hiK  = lane >> 4;

  v8f zero = {0.f, 0.f, 0.f, 0.f, 0.f, 0.f, 0.f, 0.f};
  v8f acc[4];
#pragma unroll
  for (int j = 0; j < 4; ++j) acc[j] = zero;

#if USE_TDM
  if (wave == 0) tdm_stage(HT + (size_t)n0 * MSZ, ldsX);
#pragma unroll 2
  for (int t = 0; t < 32; ++t) {
    const int k0 = t * 32;
    unsigned short* cur = ldsX + (t & 1) * TILE_ELEMS;
    if (wave == 0) {
      if (t + 1 < 32) {
        tdm_stage(HT + (size_t)n0 * MSZ + (k0 + 32),
                  ldsX + ((t + 1) & 1) * TILE_ELEMS);
        __builtin_amdgcn_s_wait_tensorcnt(1);
      } else {
        __builtin_amdgcn_s_wait_tensorcnt(0);
      }
    }
    __syncthreads();
#else
  for (int t = 0; t < 32; ++t) {
    const int k0 = t * 32;
    unsigned short* cur = ldsX;
    __syncthreads();
    {
      const int nrow = tid >> 2;
      const int ch   = (tid & 3) * 8;
      uint4 d = *(const uint4*)(HT + (size_t)(n0 + nrow) * MSZ + k0 + ch);
      *(uint4*)(cur + nrow * LDS_PITCH + ch) = d;
    }
    __syncthreads();
#endif
    const size_t arow = (size_t)mA * MSZ + (size_t)k0 + (hiK ? 8 : 0);
    Frag a;
    a.q[0] = *(const uint4*)(Wb + arow);
    a.q[1] = *(const uint4*)(Wb + arow + 16);

#pragma unroll
    for (int j = 0; j < 4; ++j) {
      const unsigned short* bp =
          cur + (j * 16 + (lane & 15)) * LDS_PITCH + (hiK ? 8 : 0);
      Frag b;
      b.q[0] = *(const uint4*)(bp);
      b.q[1] = *(const uint4*)(bp + 16);
      acc[j] = wmma_bf16(a.v, b.v, acc[j]);
    }
#if USE_TDM
    __syncthreads();
#endif
  }

#pragma unroll
  for (int j = 0; j < 4; ++j) {
    const int n = n0 + j * 16 + (lane & 15);
#pragma unroll
    for (int r = 0; r < 8; ++r) {
      const int m = m16 + (hiK ? 8 : 0) + r;
      out[(size_t)m * BSZ + n] = tanhf_fast(acc[j][r]);
    }
  }
}

// ---------------------------------------------------------------------------
extern "C" void kernel_launch(void* const* d_in, const int* in_sizes, int n_in,
                              void* d_out, int out_size, void* d_ws, size_t ws_size,
                              hipStream_t stream) {
  (void)in_sizes; (void)n_in; (void)out_size; (void)ws_size;

  const float* x             = (const float*)d_in[0];
  const float* out0          = (const float*)d_in[1];
  const float* mem0          = (const float*)d_in[2];
  const float* w_inp         = (const float*)d_in[3];
  const float* w_rec_inp     = (const float*)d_in[4];
  const float* w_inpgate     = (const float*)d_in[5];
  const float* w_mem_inpgate = (const float*)d_in[6];
  const float* w_rec_inpgate = (const float*)d_in[7];
  const float* w_readgate    = (const float*)d_in[8];
  const float* w_rec_readgate= (const float*)d_in[9];
  const float* w_mem_readgate= (const float*)d_in[10];
  // d_in[11..13] = write-gate trio: dead code in the reference, skipped.
  const float* w_hid_out     = (const float*)d_in[14];

  const size_t ACT = (size_t)MSZ * BSZ;   // 4 Mi elements
  const size_t WEL = (size_t)MSZ * MSZ;   // 1 Mi elements

  unsigned short* ws = (unsigned short*)d_ws;
  unsigned short* xT     = ws;
  unsigned short* mem0T  = ws + ACT;
  unsigned short* out0T  = ws + 2 * ACT;
  unsigned short* w_inp_b          = ws + 3 * ACT + 0 * WEL;
  unsigned short* w_inpgate_b      = ws + 3 * ACT + 1 * WEL;
  unsigned short* w_readgate_b     = ws + 3 * ACT + 2 * WEL;
  unsigned short* w_mem_inpgate_b  = ws + 3 * ACT + 3 * WEL;
  unsigned short* w_mem_readgate_b = ws + 3 * ACT + 4 * WEL;
  unsigned short* w_rec_inp_b      = ws + 3 * ACT + 5 * WEL;
  unsigned short* w_rec_inpgate_b  = ws + 3 * ACT + 6 * WEL;
  unsigned short* w_rec_readgate_b = ws + 3 * ACT + 7 * WEL;
  unsigned short* w_hid_out_b      = ws + 3 * ACT + 8 * WEL;
  unsigned short* hiddenT          = ws + 3 * ACT + 9 * WEL;

  const int TPB = 256;
  const int wG  = (int)((WEL / 4 + TPB - 1) / TPB);

  // --- 1) conversion pass: weights straight, activations transposed ---
  dim3 tGrid(BSZ / 32, MSZ / 32), tBlk(32, 8);
  hipLaunchKernelGGL(PT_GRUMB_cvt_t, tGrid, tBlk, 0, stream, x,    xT);
  hipLaunchKernelGGL(PT_GRUMB_cvt_t, tGrid, tBlk, 0, stream, mem0, mem0T);
  hipLaunchKernelGGL(PT_GRUMB_cvt_t, tGrid, tBlk, 0, stream, out0, out0T);
  hipLaunchKernelGGL(PT_GRUMB_cvt, dim3(wG), dim3(TPB), 0, stream, w_inp,          w_inp_b,          (int)(WEL / 4));
  hipLaunchKernelGGL(PT_GRUMB_cvt, dim3(wG), dim3(TPB), 0, stream, w_inpgate,      w_inpgate_b,      (int)(WEL / 4));
  hipLaunchKernelGGL(PT_GRUMB_cvt, dim3(wG), dim3(TPB), 0, stream, w_readgate,     w_readgate_b,     (int)(WEL / 4));
  hipLaunchKernelGGL(PT_GRUMB_cvt, dim3(wG), dim3(TPB), 0, stream, w_mem_inpgate,  w_mem_inpgate_b,  (int)(WEL / 4));
  hipLaunchKernelGGL(PT_GRUMB_cvt, dim3(wG), dim3(TPB), 0, stream, w_mem_readgate, w_mem_readgate_b, (int)(WEL / 4));
  hipLaunchKernelGGL(PT_GRUMB_cvt, dim3(wG), dim3(TPB), 0, stream, w_rec_inp,      w_rec_inp_b,      (int)(WEL / 4));
  hipLaunchKernelGGL(PT_GRUMB_cvt, dim3(wG), dim3(TPB), 0, stream, w_rec_inpgate,  w_rec_inpgate_b,  (int)(WEL / 4));
  hipLaunchKernelGGL(PT_GRUMB_cvt, dim3(wG), dim3(TPB), 0, stream, w_rec_readgate, w_rec_readgate_b, (int)(WEL / 4));
  hipLaunchKernelGGL(PT_GRUMB_cvt, dim3(wG), dim3(TPB), 0, stream, w_hid_out,      w_hid_out_b,      (int)(WEL / 4));

  // --- 2) fused gate GEMMs + sigmoid epilogue -> bf16 hiddenᵀ ---
  dim3 gemmGrid(BSZ / 64, MSZ / 128);
  hipLaunchKernelGGL(PT_GRUMB_gates, gemmGrid, dim3(TPB), 0, stream,
                     xT, mem0T, out0T,
                     w_inp_b, w_inpgate_b, w_readgate_b,
                     w_mem_inpgate_b, w_mem_readgate_b,
                     w_rec_inp_b, w_rec_inpgate_b, w_rec_readgate_b,
                     hiddenT);

  // --- 3) output GEMM + tanh epilogue -> f32 out ---
  hipLaunchKernelGGL(PT_GRUMB_out, gemmGrid, dim3(TPB), 0, stream,
                     w_hid_out_b, hiddenT, (float*)d_out);
}
